// Attention_19335942767108
// MI455X (gfx1250) — matrix operand order repaired and checked by reading, not compiled, not run on hardware
//
#include <hip/hip_runtime.h>
#include <hip/hip_bf16.h>

typedef __attribute__((ext_vector_type(16))) __bf16 v16bf;
typedef __attribute__((ext_vector_type(8)))  __bf16 v8bf;
typedef __attribute__((ext_vector_type(8)))  float  v8f;

union BF16x16 { v8bf h[2]; v16bf v; };

#define NPIX 65536   // 256*256
#define SDIM 256

__device__ inline v8f vzero8() { v8f z; for (int i = 0; i < 8; ++i) z[i] = 0.0f; return z; }

__device__ inline v8f wmma_bf16(const BF16x16& a, const BF16x16& b, v8f c) {
  return __builtin_amdgcn_wmma_f32_16x16x32_bf16(false, a.v, false, b.v, (short)0, c,
                                                 false, false);
}

// -----------------------------------------------------------------------------
// Kernel 1: qkv 1x1 complex conv as real GEMM.
//   T1[b][R][p] (bf16), R in [0,384): rows 0..191 real out, 192..383 imag out.
//   Wc[384][128] built on the fly: [[Wr,-Wi],[Wi,Wr]], X rows: [x_r(64); x_i(64)].
// Block: 256 thr = 8 waves. Activation tile staged ONCE (64 px x 128 K), then
// loop over 3 row-blocks of 128, re-staging only the 32KB weight tile (L2-hot).
// -----------------------------------------------------------------------------
__global__ __launch_bounds__(256) void k_qkv(const float* __restrict__ x,
                                             const float* __restrict__ qwr,
                                             const float* __restrict__ qwi,
                                             __bf16* __restrict__ T1) {
  __shared__ __bf16 sW[128][136];   // [row][K], row stride 136 (16B-aligned runs)
  __shared__ __bf16 sX[64][136];    // [px][K] transposed activations
  const int tid = threadIdx.x;
  const int p0 = blockIdx.x * 64;
  const int b  = blockIdx.z;

  // stage X tile once: rows 0..63 = x_real, 64..127 = x_imag (interleaved global)
  for (int i = 0; i < 32; ++i) {
    int idx = tid + i * 256;            // 8192 = 64*128
    int j = idx & 63, k = idx >> 6;
    float xv = x[((size_t)(b * 64 + (k & 63)) * NPIX + (p0 + j)) * 2 + (k >> 6)];
    sX[j][k] = (__bf16)xv;
  }

  const int wave = tid >> 5, lane = tid & 31;
  const int half = lane >> 4, ln = lane & 15;
  const int mbase = wave * 16;

  for (int rb = 0; rb < 3; ++rb) {
    __syncthreads();   // sX ready (rb=0); previous row-block done reading sW (rb>0)
    // stage Wc rows [rb*128, rb*128+128) as bf16
    for (int i = 0; i < 64; ++i) {
      int idx = tid + i * 256;          // 16384 = 128*128
      int c = idx & 127, r = idx >> 7;
      int R = rb * 128 + r;
      float wv;
      if (R < 192) wv = (c < 64) ? qwr[R * 64 + c] : -qwi[R * 64 + (c - 64)];
      else         wv = (c < 64) ? qwi[(R - 192) * 64 + c] : qwr[(R - 192) * 64 + (c - 64)];
      sW[r][c] = (__bf16)wv;
    }
    __syncthreads();

    v8f acc[4];
    for (int nt = 0; nt < 4; ++nt) acc[nt] = vzero8();

    for (int kk = 0; kk < 128; kk += 32) {
      BF16x16 af;                                       // A 16x32: M=ln
      const __bf16* ar = &sW[mbase + ln][0];
      af.h[0] = *reinterpret_cast<const v8bf*>(ar + kk + half * 8);        // K: kk+h*8..
      af.h[1] = *reinterpret_cast<const v8bf*>(ar + kk + 16 + half * 8);   // K: kk+16+h*8..
      for (int nt = 0; nt < 4; ++nt) {
        BF16x16 bf;                                     // B 32x16: N=ln
        const __bf16* br = &sX[nt * 16 + ln][0];
        bf.h[0] = *reinterpret_cast<const v8bf*>(br + kk + half * 8);
        bf.h[1] = *reinterpret_cast<const v8bf*>(br + kk + 16 + half * 8);
        acc[nt] = wmma_bf16(af, bf, acc[nt]);
      }
    }
    for (int nt = 0; nt < 4; ++nt)
      for (int r = 0; r < 8; ++r) {
        int R = rb * 128 + mbase + 8 * half + r;
        T1[(size_t)(b * 384 + R) * NPIX + (p0 + nt * 16 + ln)] = (__bf16)acc[nt][r];
      }
  }
}

// -----------------------------------------------------------------------------
// Kernel 2: depthwise 3x3 complex conv + sum-of-squares for q/k norms.
//   T2[b][d][ch][p] bf16. normsq[((b*2+d)*2+isK)*64 + c].
// -----------------------------------------------------------------------------
__global__ __launch_bounds__(256) void k_dw(const __bf16* __restrict__ T1,
                                            const float* __restrict__ dwr,
                                            const float* __restrict__ dwi,
                                            __bf16* __restrict__ T2,
                                            float* __restrict__ normsq) {
  __shared__ float red[256];
  const int tid = threadIdx.x;
  const int p  = blockIdx.x * 256 + tid;
  const int ch = blockIdx.y;
  const int b  = blockIdx.z;
  const int s0 = p >> 8, s1 = p & 255;
  const __bf16* inR = T1 + ((size_t)(b * 384) + ch) * NPIX;
  const __bf16* inI = T1 + ((size_t)(b * 384) + 192 + ch) * NPIX;
  const float* wr = dwr + ch * 9;
  const float* wi = dwi + ch * 9;

  // hint the next row of both planes into cache (global_prefetch_b8)
  if (s0 + 1 < SDIM) {
    __builtin_prefetch(inR + (s0 + 1) * SDIM + s1, 0, 1);
    __builtin_prefetch(inI + (s0 + 1) * SDIM + s1, 0, 1);
  }

  float outr = 0.0f, outi = 0.0f;
  for (int dy = -1; dy <= 1; ++dy) {
    int y = s0 + dy;
    if ((unsigned)y >= (unsigned)SDIM) continue;
    for (int dx = -1; dx <= 1; ++dx) {
      int xs = s1 + dx;
      if ((unsigned)xs >= (unsigned)SDIM) continue;
      int q = y * SDIM + xs;
      float ar = (float)inR[q], ai = (float)inI[q];
      float wrv = wr[(dy + 1) * 3 + (dx + 1)];
      float wiv = wi[(dy + 1) * 3 + (dx + 1)];
      outr += wrv * ar - wiv * ai;
      outi += wrv * ai + wiv * ar;
    }
  }
  T2[((size_t)(b * 2 + 0) * 192 + ch) * NPIX + p] = (__bf16)outr;
  T2[((size_t)(b * 2 + 1) * 192 + ch) * NPIX + p] = (__bf16)outi;

  if (ch < 128) {   // q: ch<64, k: 64..127 — accumulate squared norms per part d
    int isK = ch >> 6, cc = ch & 63;
    red[tid] = outr * outr;
    __syncthreads();
    for (int s = 128; s > 0; s >>= 1) {
      if (tid < s) red[tid] += red[tid + s];
      __syncthreads();
    }
    if (tid == 0) atomicAdd(&normsq[((b * 2 + 0) * 2 + isK) * 64 + cc], red[0]);
    __syncthreads();
    red[tid] = outi * outi;
    __syncthreads();
    for (int s = 128; s > 0; s >>= 1) {
      if (tid < s) red[tid] += red[tid + s];
      __syncthreads();
    }
    if (tid == 0) atomicAdd(&normsq[((b * 2 + 1) * 2 + isK) * 64 + cc], red[0]);
  }
}

// -----------------------------------------------------------------------------
// Kernel 3: S[b][d][64][64] = q . k^T (unnormalized), split-K over 65536 tokens.
// Each wave: 512 tokens, full 64x64 output (16 accum tiles), atomicAdd merge.
// Fragments loaded directly from global with 16B vector loads (token runs are
// contiguous for both A and B per the documented VGPR layouts).
// -----------------------------------------------------------------------------
__global__ __launch_bounds__(256) void k_attn(const __bf16* __restrict__ T2,
                                              float* __restrict__ S) {
  const int tid = threadIdx.x, wave = tid >> 5, lane = tid & 31;
  const int half = lane >> 4, ln = lane & 15;
  const int d = blockIdx.y, b = blockIdx.z;
  const __bf16* Q = T2 + (size_t)((b * 2 + d) * 192 + 0) * NPIX;
  const __bf16* K = T2 + (size_t)((b * 2 + d) * 192 + 64) * NPIX;
  const int tok0 = (blockIdx.x * 8 + wave) * 512;

  v8f acc[4][4];
  for (int mt = 0; mt < 4; ++mt)
    for (int nt = 0; nt < 4; ++nt) acc[mt][nt] = vzero8();

  for (int it = 0; it < 16; ++it) {
    int t0 = tok0 + it * 32;
    BF16x16 bf[4];
    for (int nt = 0; nt < 4; ++nt) {
      const __bf16* kr = K + (size_t)(nt * 16 + ln) * NPIX + t0 + half * 8;
      bf[nt].h[0] = *reinterpret_cast<const v8bf*>(kr);
      bf[nt].h[1] = *reinterpret_cast<const v8bf*>(kr + 16);
    }
    for (int mt = 0; mt < 4; ++mt) {
      BF16x16 af;
      const __bf16* qr = Q + (size_t)(mt * 16 + ln) * NPIX + t0;
      af.h[0] = *reinterpret_cast<const v8bf*>(qr + half * 8);
      af.h[1] = *reinterpret_cast<const v8bf*>(qr + 16 + half * 8);
      for (int nt = 0; nt < 4; ++nt) acc[mt][nt] = wmma_bf16(af, bf[nt], acc[mt][nt]);
    }
  }
  float* Sp = S + (size_t)((b * 2 + d) * 64 * 64);
  for (int mt = 0; mt < 4; ++mt)
    for (int nt = 0; nt < 4; ++nt)
      for (int r = 0; r < 8; ++r)
        atomicAdd(&Sp[(mt * 16 + 8 * half + r) * 64 + nt * 16 + ln], acc[mt][nt][r]);
}

// -----------------------------------------------------------------------------
// Kernel 4: fold cosine normalization into S, row softmax -> P. One thread/row.
// -----------------------------------------------------------------------------
__global__ __launch_bounds__(64) void k_softmax(const float* __restrict__ S,
                                                const float* __restrict__ normsq,
                                                float* __restrict__ P) {
  const int c = threadIdx.x;                  // 0..63 (row)
  const int d = blockIdx.x & 1, b = blockIdx.x >> 1;
  const float* Sp  = S + (size_t)((b * 2 + d) * 64 * 64);
  float*       Pp  = P + (size_t)((b * 2 + d) * 64 * 64);
  const float* nq2 = normsq + ((b * 2 + d) * 2 + 0) * 64;
  const float* nk2 = normsq + ((b * 2 + d) * 2 + 1) * 64;
  const float eps = 1e-12f;
  float inq = 1.0f / fmaxf(sqrtf(nq2[c]), eps);
  float vals[64];
  float mx = -3.4e38f;
  for (int e = 0; e < 64; ++e) {
    float v = Sp[c * 64 + e] * inq / fmaxf(sqrtf(nk2[e]), eps);
    vals[e] = v;
    mx = fmaxf(mx, v);
  }
  float sum = 0.0f;
  for (int e = 0; e < 64; ++e) { float ev = __expf(vals[e] - mx); vals[e] = ev; sum += ev; }
  float inv = 1.0f / sum;
  for (int e = 0; e < 64; ++e) Pp[c * 64 + e] = vals[e] * inv;
}

// -----------------------------------------------------------------------------
// Kernel 5: M_b[128][128] = Wproj_complex . blockdiag(P0, P1)   (per batch).
// Fuses attention-apply and projection into one downstream GEMM.
// -----------------------------------------------------------------------------
__global__ __launch_bounds__(256) void k_mbuild(const float* __restrict__ pwr,
                                                const float* __restrict__ pwi,
                                                const float* __restrict__ P,
                                                float* __restrict__ Mb) {
  const int b = blockIdx.x, tid = threadIdx.x;
  const float* P0 = P + (size_t)(b * 2 + 0) * 4096;
  const float* P1 = P + (size_t)(b * 2 + 1) * 4096;
  for (int i = 0; i < 64; ++i) {
    int idx = tid + i * 256;                  // 16384
    int col = idx & 127, r = idx >> 7;
    float s = 0.0f;
    if (col < 64) {
      for (int c = 0; c < 64; ++c) {
        float w = (r < 64) ? pwr[r * 64 + c] : pwi[(r - 64) * 64 + c];
        s += w * P0[c * 64 + col];
      }
    } else {
      int e = col - 64;
      for (int c = 0; c < 64; ++c) {
        float w = (r < 64) ? -pwi[r * 64 + c] : pwr[(r - 64) * 64 + c];
        s += w * P1[c * 64 + e];
      }
    }
    Mb[(size_t)b * 16384 + r * 128 + col] = s;
  }
}

// -----------------------------------------------------------------------------
// Kernel 6: out[b][c][p][d] = M_b . V[b][:, p]  — WMMA GEMM, same shape as k_qkv.
// V rows: 0..63 = v_real (T2 d=0 ch 128..191), 64..127 = v_imag (d=1).
// -----------------------------------------------------------------------------
__global__ __launch_bounds__(256) void k_out(const __bf16* __restrict__ T2,
                                             const float* __restrict__ Mb,
                                             float* __restrict__ out) {
  __shared__ __bf16 sW[128][136];
  __shared__ __bf16 sX[64][136];
  const int tid = threadIdx.x;
  const int p0 = blockIdx.x * 64;
  const int b  = blockIdx.z;

  for (int i = 0; i < 64; ++i) {
    int idx = tid + i * 256;
    int c = idx & 127, r = idx >> 7;
    sW[r][c] = (__bf16)Mb[(size_t)b * 16384 + idx];
  }
  for (int i = 0; i < 32; ++i) {
    int idx = tid + i * 256;
    int j = idx & 63, k = idx >> 6;
    sX[j][k] = T2[((size_t)(b * 2 + (k >> 6)) * 192 + 128 + (k & 63)) * NPIX + p0 + j];
  }
  __syncthreads();

  const int wave = tid >> 5, lane = tid & 31;
  const int half = lane >> 4, ln = lane & 15;
  const int mbase = wave * 16;

  v8f acc[4];
  for (int nt = 0; nt < 4; ++nt) acc[nt] = vzero8();

  for (int kk = 0; kk < 128; kk += 32) {
    BF16x16 af;
    const __bf16* ar = &sW[mbase + ln][0];
    af.h[0] = *reinterpret_cast<const v8bf*>(ar + kk + half * 8);
    af.h[1] = *reinterpret_cast<const v8bf*>(ar + kk + 16 + half * 8);
    for (int nt = 0; nt < 4; ++nt) {
      BF16x16 bf;
      const __bf16* br = &sX[nt * 16 + ln][0];
      bf.h[0] = *reinterpret_cast<const v8bf*>(br + kk + half * 8);
      bf.h[1] = *reinterpret_cast<const v8bf*>(br + kk + 16 + half * 8);
      acc[nt] = wmma_bf16(af, bf, acc[nt]);
    }
  }
  for (int nt = 0; nt < 4; ++nt)
    for (int r = 0; r < 8; ++r) {
      int R = mbase + 8 * half + r;        // 0..127
      int dd = R >> 6, cch = R & 63;
      size_t p = (size_t)(p0 + nt * 16 + ln);
      out[(((size_t)(b * 64 + cch)) * NPIX + p) * 2 + dd] = acc[nt][r];
    }
}

// -----------------------------------------------------------------------------
extern "C" void kernel_launch(void* const* d_in, const int* in_sizes, int n_in,
                              void* d_out, int out_size, void* d_ws, size_t ws_size,
                              hipStream_t stream) {
  (void)in_sizes; (void)n_in; (void)out_size; (void)ws_size;
  const float* x       = (const float*)d_in[0];
  const float* qkv_wr  = (const float*)d_in[1];
  const float* qkv_wi  = (const float*)d_in[2];
  const float* dw_wr   = (const float*)d_in[3];
  const float* dw_wi   = (const float*)d_in[4];
  const float* proj_wr = (const float*)d_in[5];
  const float* proj_wi = (const float*)d_in[6];

  char* ws = (char*)d_ws;
  const size_t T1_BYTES = (size_t)2 * 384 * NPIX * 2;   // bf16, ~96 MiB
  const size_t T2_BYTES = (size_t)2 * 2 * 192 * NPIX * 2;
  __bf16* T1     = (__bf16*)ws;                 ws += T1_BYTES;
  __bf16* T2     = (__bf16*)ws;                 ws += T2_BYTES;
  float*  normsq = (float*)ws;                  ws += 1024 * sizeof(float);
  float*  S      = (float*)ws;                  ws += 16384 * sizeof(float);
  float*  P      = (float*)ws;                  ws += 16384 * sizeof(float);
  float*  Mb     = (float*)ws;

  // zero split-K / norm accumulators (normsq and S are contiguous)
  hipMemsetAsync(normsq, 0, (1024 + 16384) * sizeof(float), stream);

  dim3 blk(256);
  k_qkv    <<<dim3(NPIX / 64, 1, 2), blk, 0, stream>>>(x, qkv_wr, qkv_wi, T1);
  k_dw     <<<dim3(NPIX / 256, 192, 2), blk, 0, stream>>>(T1, dw_wr, dw_wi, T2, normsq);
  k_attn   <<<dim3(16, 2, 2), blk, 0, stream>>>(T2, S);
  k_softmax<<<dim3(4), dim3(64), 0, stream>>>(S, normsq, P);
  k_mbuild <<<dim3(2), blk, 0, stream>>>(proj_wr, proj_wi, P, Mb);
  k_out    <<<dim3(NPIX / 64, 1, 2), blk, 0, stream>>>(T2, Mb, (float*)d_out);
}
